// ScaledDotProductAttentionWithRelative_61735859912865
// MI455X (gfx1250) — compile-verified, matrix operand order checked by reading, not run on hardware
//
#include <hip/hip_runtime.h>
#include <math.h>

typedef __attribute__((ext_vector_type(16))) _Float16 v16h;
typedef __attribute__((ext_vector_type(4)))  _Float16 v4h;
typedef __attribute__((ext_vector_type(8)))  float    v8f;

#define D_K      128
#define L_SEQ    1024
#define BLOCK_M  128     // rows per workgroup = 8 waves x 16 rows
#define NWAVES   8
#define VT_LD    40      // transposed-V LDS row stride (halves); 80B keeps 16B alignment

// Element e (0..15) of a 16-bit A/B fragment (v16h), lane half g (0/1):
// K offset within the 32-wide slab per the CDNA5 WMMA VGPR layout.
__device__ __forceinline__ int kmap(int e, int g) {
    return ((e & 8) ? (e + 8) : e) + (g << 3);
}

__global__ __launch_bounds__(256)
void attn_rel_kernel(const float* __restrict__ Q,
                     const float* __restrict__ K,
                     const float* __restrict__ V,
                     const float* __restrict__ REL,
                     const int*   __restrict__ MASK,
                     float* __restrict__ OUT,    // [B,H,L,D]
                     float* __restrict__ ATTN)   // [B,H,L,L]
{
    const int bh  = blockIdx.y;            // 0..63  (b*16 + h)
    const int b   = bh >> 4;               // H = 16
    const int tid = threadIdx.x;
    const int wave = tid >> 5;
    const int lane = tid & 31;
    const int n = lane & 15;               // column / row-in-tile index
    const int g = lane >> 4;               // lane half

    const float scale = 0.08838834764831845f;   // 1/sqrt(128)

    const size_t bh_off = (size_t)bh * L_SEQ * D_K;
    const float* Qb = Q   + bh_off;
    const float* Kb = K   + bh_off;
    const float* Vb = V   + bh_off;
    const float* Rb = REL + bh_off;
    const int*   Mb = MASK + (size_t)b * L_SEQ * L_SEQ;
    float* Ob = OUT  + bh_off;
    float* Ab = ATTN + (size_t)bh * L_SEQ * L_SEQ;

    const int r0 = blockIdx.x * BLOCK_M + wave * 16;   // wave's first Q row

    __shared__ _Float16 sKp[2][32 * D_K];          // K' = K*scale + rel (double-buffered)
    __shared__ _Float16 sVt[2][D_K * VT_LD];       // V transposed [dcol][k] (double-buffered)
    __shared__ _Float16 sP [NWAVES * 16 * 32];     // per-wave P staging (16x32)

    // ---- stage helpers -------------------------------------------------
    // Pass-1 tile: 16 rows of K' (rows j*16..+15), float4 per thread chunk.
    auto stage1 = [&](int j, int buf) {
        #pragma unroll
        for (int i4 = 0; i4 < (16 * D_K) / (4 * 256); ++i4) {
            int idx = (tid + i4 * 256) * 4;
            int rr = idx >> 7, cc = idx & 127;
            const float4 kv = *(const float4*)(Kb + (size_t)(j * 16 + rr) * D_K + cc);
            const float4 rv = *(const float4*)(Rb + (size_t)(j * 16 + rr) * D_K + cc);
            v4h h;
            h[0] = (_Float16)fmaf(scale, kv.x, rv.x);
            h[1] = (_Float16)fmaf(scale, kv.y, rv.y);
            h[2] = (_Float16)fmaf(scale, kv.z, rv.z);
            h[3] = (_Float16)fmaf(scale, kv.w, rv.w);
            *(v4h*)(&sKp[buf][idx]) = h;           // 8B-aligned -> ds_store_b64
        }
    };
    // Pass-2 tile: 32 rows of K' and V (rows jp*32..+31); V stored transposed.
    auto stage2 = [&](int jp, int buf) {
        #pragma unroll
        for (int i4 = 0; i4 < (32 * D_K) / (4 * 256); ++i4) {
            int idx = (tid + i4 * 256) * 4;
            int rr = idx >> 7, cc = idx & 127;
            size_t gi = (size_t)(jp * 32 + rr) * D_K + cc;
            const float4 kv = *(const float4*)(Kb + gi);
            const float4 rv = *(const float4*)(Rb + gi);
            const float4 vv = *(const float4*)(Vb + gi);
            v4h h;
            h[0] = (_Float16)fmaf(scale, kv.x, rv.x);
            h[1] = (_Float16)fmaf(scale, kv.y, rv.y);
            h[2] = (_Float16)fmaf(scale, kv.z, rv.z);
            h[3] = (_Float16)fmaf(scale, kv.w, rv.w);
            *(v4h*)(&sKp[buf][idx]) = h;
            sVt[buf][(cc + 0) * VT_LD + rr] = (_Float16)vv.x;   // transpose scatter
            sVt[buf][(cc + 1) * VT_LD + rr] = (_Float16)vv.y;
            sVt[buf][(cc + 2) * VT_LD + rr] = (_Float16)vv.z;
            sVt[buf][(cc + 3) * VT_LD + rr] = (_Float16)vv.w;
        }
    };

    // ---- Q fragments (A layout), resident: 16x128 = 4 slabs of K=32 ----
    v16h qf[4];
    {
        const float* qrow = Qb + (size_t)(r0 + n) * D_K;
        #pragma unroll
        for (int kk = 0; kk < 4; ++kk) {
            const int base = kk * 32 + 8 * g;
            const float4 a = *(const float4*)(qrow + base);
            const float4 bq = *(const float4*)(qrow + base + 4);
            const float4 cq = *(const float4*)(qrow + base + 16);
            const float4 d = *(const float4*)(qrow + base + 20);
            v16h f;
            f[0] = (_Float16)a.x;  f[1] = (_Float16)a.y;  f[2] = (_Float16)a.z;  f[3] = (_Float16)a.w;
            f[4] = (_Float16)bq.x; f[5] = (_Float16)bq.y; f[6] = (_Float16)bq.z; f[7] = (_Float16)bq.w;
            f[8] = (_Float16)cq.x; f[9] = (_Float16)cq.y; f[10] = (_Float16)cq.z; f[11] = (_Float16)cq.w;
            f[12] = (_Float16)d.x; f[13] = (_Float16)d.y; f[14] = (_Float16)d.z; f[15] = (_Float16)d.w;
            qf[kk] = f;
        }
    }

    const v8f vzero = {};
    v8f oacc[8];
    #pragma unroll
    for (int t = 0; t < 8; ++t) oacc[t] = vzero;

    float sumloc[8];
    #pragma unroll
    for (int v = 0; v < 8; ++v) sumloc[v] = 0.0f;

    // ================= PASS 1: row sums of exp(masked scores) =================
    stage1(0, 0);
    __syncthreads();
    for (int j = 0; j < L_SEQ / 16; ++j) {
        if (j + 1 < L_SEQ / 16) stage1(j + 1, (j + 1) & 1);    // overlap next tile
        if (j + 2 < L_SEQ / 16) {                              // L2 prefetch 2 ahead
            size_t pg = (size_t)((j + 2) * 16 + (tid >> 4)) * D_K + (tid & 15) * 8;
            __builtin_prefetch(Kb + pg, 0, 1);
            __builtin_prefetch(Rb + pg, 0, 1);
        }
        const _Float16* kp = sKp[j & 1];
        v8f c = vzero;
        #pragma unroll
        for (int kk = 0; kk < 4; ++kk) {
            v16h bf;                               // B fragment: column n = K' row n
            #pragma unroll
            for (int e = 0; e < 16; ++e)
                bf[e] = kp[n * D_K + kk * 32 + kmap(e, g)];
            c = __builtin_amdgcn_wmma_f32_16x16x32_f16(false, qf[kk], false, bf,
                                                       (short)0, c, false, false);
        }
        #pragma unroll
        for (int v = 0; v < 8; ++v) {
            int row = r0 + v + 8 * g;
            int col = j * 16 + n;
            float s = (Mb[(size_t)row * L_SEQ + col] == 0) ? -1e9f : c[v];
            sumloc[v] += __expf(s);                // exp(-1e9) == 0
        }
        __syncthreads();
    }
    // reduce partial sums across the 16 lanes of each half (rows stay separate)
    #pragma unroll
    for (int v = 0; v < 8; ++v) {
        float s = sumloc[v];
        #pragma unroll
        for (int m = 1; m < 16; m <<= 1) s += __shfl_xor(s, m, 32);
        sumloc[v] = 1.0f / s;                      // inverse row sum
    }
    __syncthreads();

    // ====== PASS 2: recompute S, write normalized attention, O += P*V ======
    stage2(0, 0);
    __syncthreads();
    for (int jp = 0; jp < L_SEQ / 32; ++jp) {
        if (jp + 1 < L_SEQ / 32) stage2(jp + 1, (jp + 1) & 1);
        if (jp + 2 < L_SEQ / 32) {
            size_t pg = (size_t)((jp + 2) * 32 + (tid >> 3)) * D_K + (tid & 7) * 16;
            __builtin_prefetch(Kb + pg, 0, 1);
            __builtin_prefetch(Rb + pg, 0, 1);
            __builtin_prefetch(Vb + pg, 0, 1);
        }
        const int buf = jp & 1;
        const _Float16* kp = sKp[buf];
        const _Float16* vt = sVt[buf];
        _Float16* myP = sP + wave * (16 * 32);

        #pragma unroll
        for (int sub = 0; sub < 2; ++sub) {
            v8f c = vzero;
            #pragma unroll
            for (int kk = 0; kk < 4; ++kk) {
                v16h bf;
                #pragma unroll
                for (int e = 0; e < 16; ++e)
                    bf[e] = kp[(sub * 16 + n) * D_K + kk * 32 + kmap(e, g)];
                c = __builtin_amdgcn_wmma_f32_16x16x32_f16(false, qf[kk], false, bf,
                                                           (short)0, c, false, false);
            }
            #pragma unroll
            for (int v = 0; v < 8; ++v) {
                int rloc = v + 8 * g;
                int row  = r0 + rloc;
                int col  = jp * 32 + sub * 16 + n;
                float s  = (Mb[(size_t)row * L_SEQ + col] == 0) ? -1e9f : c[v];
                float p  = __expf(s) * sumloc[v];
                Ab[(size_t)row * L_SEQ + col] = p;           // sole attention write
                myP[rloc * 32 + sub * 16 + n] = (_Float16)p;
            }
        }
        // P staging is wave-private: a wave's DS ops complete in order; dscnt 0
        // guarantees the b16 stores landed before the b128 fragment reads.
        asm volatile("s_wait_dscnt 0x0" ::: "memory");

        v16h pf;                                   // A fragment of P (16x32)
        #pragma unroll
        for (int e = 0; e < 16; ++e)
            pf[e] = myP[n * 32 + kmap(e, g)];
        #pragma unroll
        for (int t = 0; t < 8; ++t) {
            v16h vf;                               // B fragment from transposed V
            #pragma unroll
            for (int e = 0; e < 16; ++e)
                vf[e] = vt[(t * 16 + n) * VT_LD + kmap(e, g)];
            oacc[t] = __builtin_amdgcn_wmma_f32_16x16x32_f16(false, pf, false, vf,
                                                             (short)0, oacc[t], false, false);
        }
        __syncthreads();   // next-tile staging done; all reads of current buf done
    }

    // ---- write O (C layout: lane n = col, vgpr v + 8g = row)
    #pragma unroll
    for (int t = 0; t < 8; ++t) {
        #pragma unroll
        for (int v = 0; v < 8; ++v) {
            int row = r0 + v + 8 * g;
            Ob[(size_t)row * D_K + t * 16 + n] = oacc[t][v];
        }
    }
}

extern "C" void kernel_launch(void* const* d_in, const int* in_sizes, int n_in,
                              void* d_out, int out_size, void* d_ws, size_t ws_size,
                              hipStream_t stream) {
    const float* Q   = (const float*)d_in[0];
    const float* K   = (const float*)d_in[1];
    const float* V   = (const float*)d_in[2];
    const float* REL = (const float*)d_in[3];
    const int*   M   = (const int*)d_in[4];

    float* OUT  = (float*)d_out;                              // [4,16,1024,128]
    float* ATTN = OUT + (size_t)4 * 16 * 1024 * 128;          // [4,16,1024,1024]

    dim3 grid(L_SEQ / BLOCK_M, 4 * 16);   // 8 row-blocks x 64 heads
    attn_rel_kernel<<<grid, 256, 0, stream>>>(Q, K, V, REL, M, OUT, ATTN);
}
